// BoxMultiHeadedAttention_5196910428328
// MI455X (gfx1250) — matrix operand order repaired and checked by reading, not compiled, not run on hardware
//
#include <hip/hip_runtime.h>
#include <hip/hip_bf16.h>

// ---------------------------------------------------------------------------
// BoxMultiHeadedAttention for MI455X (gfx1250, wave32, WMMA)
//   B=16, N=256, D=1024, H=8, d_k=128, DIM_G=64
// Pipeline:
//   0) prep_weights : W[k][n] fp32 -> Wt[n][k] f16 (4 weights)
//   1) gemm_proj x3 : X @ W + b -> Q/K [B,H,N,dk] f16, V [B,H,dk,N] f16 (WMMA)
//   2) bias_kernel  : fused box-geometry -> log(relu(emb.Wg+bg)) [B,H,N,N] f16
//   3) attn_kernel  : fused scores(WMMA) + bias + softmax + P.V(WMMA) -> f16
//   4) gemm_proj    : attn @ Wo + bo -> fp32 d_out (WMMA)
// gfx1250 features: v_wmma_f32_16x16x32_f16, global_load_async_to_lds_b128
// (ASYNCcnt) for pure f16 tile staging, global_prefetch_b8, split barriers.
// ---------------------------------------------------------------------------

typedef __attribute__((ext_vector_type(16))) _Float16 v16h;
typedef __attribute__((ext_vector_type(8)))  float    v8f;

union H16 {
    v16h     v;
    _Float16 h[16];
    int4     q[2];
};

#define BB   16
#define NN   256
#define DD   1024
#define HH   8
#define DK   128

// ---------------------------------------------------------------------------
// Kernel 0: weight convert + transpose.  Wt[n*1024 + k] = (f16) W[k*1024 + n]
// ---------------------------------------------------------------------------
__global__ __launch_bounds__(256) void prep_weights(
    const float* __restrict__ Wq, const float* __restrict__ Wk,
    const float* __restrict__ Wv, const float* __restrict__ Wo,
    _Float16* __restrict__ out)
{
    const float* W = (blockIdx.y == 0) ? Wq : (blockIdx.y == 1) ? Wk
                   : (blockIdx.y == 2) ? Wv : Wo;
    size_t idx = (size_t)blockIdx.x * 256 + threadIdx.x;   // 0 .. 1M-1
    int n = (int)(idx >> 10);
    int k = (int)(idx & 1023);
    out[(size_t)blockIdx.y * (1u << 20) + idx] = (_Float16)W[(size_t)k * DD + n];
}

// ---------------------------------------------------------------------------
// Kernel 1/4: 4096x1024x1024 GEMM with f16 WMMA.
//   A: fp32 (projections) or f16 (output proj), row-major [4096,1024]
//   Wt: f16, [1024(out col), 1024(k)] row-major (pre-transposed)
//   mode 0/1: out f16 [B,H,N,dk]   (Q / K)
//   mode 2  : out f16 [B,H,dk,N]   (V transposed)
//   mode 3  : out fp32 [4096,1024] (final output, + bias)
// Block: 256 threads = 8 waves. Block tile 128x128, K-step 32.
// f16 tiles are staged with GLOBAL_LOAD_ASYNC_TO_LDS_B128 (ASYNCcnt);
// fp32 A tiles go through VGPRs for the f32->f16 convert.
// ---------------------------------------------------------------------------
__global__ __launch_bounds__(256) void gemm_proj(
    const float* __restrict__ A32, const _Float16* __restrict__ A16,
    const _Float16* __restrict__ Wt, const float* __restrict__ bias,
    _Float16* __restrict__ out_h, float* __restrict__ out_f, int mode)
{
    __shared__ alignas(16) _Float16 sA[128][48];   // 32 used + pad
    __shared__ alignas(16) _Float16 sB[128][48];

    const int tid  = threadIdx.x;
    const int wave = tid >> 5;
    const int lane = tid & 31;
    const int lm   = lane & 15;
    const int lh   = lane >> 4;

    const int ldRow = tid >> 1;          // 0..127
    const int ldSeg = (tid & 1) * 16;    // 0 or 16

    const int blockM = blockIdx.x * 128;
    const int blockN = blockIdx.y * 128;

    v8f acc[8];
    const v8f vzero = {0.f, 0.f, 0.f, 0.f, 0.f, 0.f, 0.f, 0.f};
#pragma unroll
    for (int ct = 0; ct < 8; ++ct) acc[ct] = vzero;

    for (int kk = 0; kk < 32; ++kk) {
        const int kbase = kk * 32;

        // ---- B tile (128 cols x 32 k): async global -> LDS, no VGPR round-trip
        {
            const _Float16* p = Wt + (size_t)(blockN + ldRow) * DD + kbase + ldSeg;
            unsigned lb = (unsigned)(size_t)&sB[ldRow][ldSeg];
            // INST_OFFSET is added to BOTH the LDS and global addresses.
            asm volatile(
                "global_load_async_to_lds_b128 %0, %1, off\n\t"
                "global_load_async_to_lds_b128 %0, %1, off offset:16"
                :: "v"(lb), "v"(p) : "memory");
            if (kk < 31) __builtin_prefetch(p + 32, 0, 1);
        }

        // ---- A tile (128 rows x 32 k) ----
        if (A32) {   // fp32 source: load, convert, ds_store
            H16 u;
            const float* p = A32 + (size_t)(blockM + ldRow) * DD + kbase + ldSeg;
#pragma unroll
            for (int t = 0; t < 16; ++t) u.h[t] = (_Float16)p[t];
            if (kk < 31) __builtin_prefetch(p + 32, 0, 1);
            *(int4*)&sA[ldRow][ldSeg]     = u.q[0];
            *(int4*)&sA[ldRow][ldSeg + 8] = u.q[1];
        } else {     // f16 source: async direct to LDS
            const _Float16* p = A16 + (size_t)(blockM + ldRow) * DD + kbase + ldSeg;
            unsigned la = (unsigned)(size_t)&sA[ldRow][ldSeg];
            asm volatile(
                "global_load_async_to_lds_b128 %0, %1, off\n\t"
                "global_load_async_to_lds_b128 %0, %1, off offset:16"
                :: "v"(la), "v"(p) : "memory");
            if (kk < 31) __builtin_prefetch(p + 32, 0, 1);
        }

        // All of this thread's async LDS writes done, then block-wide barrier.
        asm volatile("s_wait_asynccnt 0x0" ::: "memory");
        __syncthreads();

        // ---- A fragment: rows 16*wave .. +15 ----
        H16 fa;
        {
            const int r  = wave * 16 + lm;
            const int ko = lh * 16;
            fa.q[0] = *(const int4*)&sA[r][ko];
            fa.q[1] = *(const int4*)&sA[r][ko + 8];
        }
        // ---- batch all 8 B fragments, then chain the WMMAs ----
        H16 fb[8];
#pragma unroll
        for (int ct = 0; ct < 8; ++ct) {
            const int c  = ct * 16 + lm;
            const int kb = lh * 16;
            fb[ct].q[0] = *(const int4*)&sB[c][kb];
            fb[ct].q[1] = *(const int4*)&sB[c][kb + 8];
        }
#pragma unroll
        for (int ct = 0; ct < 8; ++ct)
            acc[ct] = __builtin_amdgcn_wmma_f32_16x16x32_f16(
                false, fa.v, false, fb[ct].v, (short)0, acc[ct], false, false);
        __syncthreads();
    }

    // ---- epilogue ----
#pragma unroll
    for (int ct = 0; ct < 8; ++ct) {
        const int gcol = blockN + ct * 16 + lm;
        const float bv = bias[gcol];
#pragma unroll
        for (int i = 0; i < 8; ++i) {
            const int grow = blockM + wave * 16 + lh * 8 + i;
            const float val = acc[ct][i] + bv;
            if (mode == 3) {
                out_f[(size_t)grow * DD + gcol] = val;
            } else {
                const int b = grow >> 8, n = grow & 255;
                const int h = gcol >> 7, d = gcol & 127;
                if (mode == 2)   // V transposed: [B,H,dk,N]
                    out_h[(((size_t)b * HH + h) * DK + d) * NN + n] = (_Float16)val;
                else             // Q / K: [B,H,N,dk]
                    out_h[(((size_t)b * HH + h) * NN + n) * DK + d] = (_Float16)val;
            }
        }
    }
}

// ---------------------------------------------------------------------------
// Kernel 2: fused box relational embedding -> log(relu(emb.Wg + bg)) f16.
// One block per (b, i); 256 threads = j. Never materializes emb.
// ---------------------------------------------------------------------------
__global__ __launch_bounds__(256) void bias_kernel(
    const float* __restrict__ box, const float* __restrict__ Wg,
    const float* __restrict__ bg, _Float16* __restrict__ biasL)
{
    __shared__ float sWg[512];
    __shared__ float sbg[8];
    const int j = threadIdx.x;
    const int i = blockIdx.x;
    const int b = blockIdx.y;

    sWg[j]       = Wg[j];
    sWg[j + 256] = Wg[j + 256];
    if (j < 8) sbg[j] = bg[j];
    __syncthreads();

    const float* bi = box + ((size_t)b * NN + i) * 4;
    const float* bj = box + ((size_t)b * NN + j) * 4;
    const float cxi = (bi[0] + bi[2]) * 0.5f, cyi = (bi[1] + bi[3]) * 0.5f;
    const float wi  = bi[2] - bi[0] + 1.0f,   hi  = bi[3] - bi[1] + 1.0f;
    const float cxj = (bj[0] + bj[2]) * 0.5f, cyj = (bj[1] + bj[3]) * 0.5f;
    const float wj  = bj[2] - bj[0] + 1.0f,   hj  = bj[3] - bj[1] + 1.0f;

    float pos[4];
    pos[0] = __logf(fmaxf(fabsf((cxi - cxj) / wi), 1e-3f));
    pos[1] = __logf(fmaxf(fabsf((cyi - cyj) / hi), 1e-3f));
    pos[2] = __logf(wi / wj);
    pos[3] = __logf(hi / hj);

    const float dm[8] = {1.0f, 0.421696503f, 0.177827941f, 0.074989421f,
                         0.031622777f, 0.013335214f, 0.005623413f, 0.002371374f};

    float acch[8] = {0.f, 0.f, 0.f, 0.f, 0.f, 0.f, 0.f, 0.f};
#pragma unroll
    for (int c = 0; c < 4; ++c) {
        const float base = 100.0f * pos[c];
#pragma unroll
        for (int r = 0; r < 8; ++r) {
            float sn, cs;
            __sincosf(base * dm[r], &sn, &cs);
            const int g = c * 8 + r;
#pragma unroll
            for (int h = 0; h < 8; ++h)
                acch[h] += sWg[h * 64 + g] * sn + sWg[h * 64 + 32 + g] * cs;
        }
    }
#pragma unroll
    for (int h = 0; h < 8; ++h) {
        const float rel = fmaxf(acch[h] + sbg[h], 0.0f);
        biasL[(((size_t)(b * HH + h) * NN) + i) * NN + j] =
            (_Float16)__logf(fmaxf(rel, 1e-6f));
    }
}

// ---------------------------------------------------------------------------
// Kernel 3: fused attention. Block = 4 waves, each wave owns a 16-row Q tile
// and the full 256-key axis (16 v8f score accumulators). WMMA for S = Q.K^T
// and O = P.V; softmax reductions via 16-lane shfl_xor matching C-layout.
// ---------------------------------------------------------------------------
__global__ __launch_bounds__(128) void attn_kernel(
    const _Float16* __restrict__ Qh, const _Float16* __restrict__ Kh,
    const _Float16* __restrict__ Vt, const _Float16* __restrict__ biasL,
    _Float16* __restrict__ Oout)
{
    __shared__ alignas(16) _Float16 pl[4][16][264];   // per-wave P staging

    const int wave = threadIdx.x >> 5;
    const int lane = threadIdx.x & 31;
    const int lm   = lane & 15;
    const int lh   = lane >> 4;

    const int bh    = blockIdx.x >> 2;          // b*8 + h
    const int qbase = (blockIdx.x & 3) * 64 + wave * 16;
    const int b     = bh >> 3;
    const int h     = bh & 7;

    const _Float16* Qb = Qh + (size_t)bh * NN * DK;
    const _Float16* Kb = Kh + (size_t)bh * NN * DK;
    const _Float16* Vb = Vt + (size_t)bh * DK * NN;
    const _Float16* Bb = biasL + (size_t)bh * NN * NN;

    // ---- Q fragments (16 x 128 = 4 K-chunks), kept in registers ----
    H16 fq[4];
    {
        const _Float16* qrow = Qb + (size_t)(qbase + lm) * DK;
        const int ko = lh * 16;
#pragma unroll
        for (int kc = 0; kc < 4; ++kc) {
            fq[kc].q[0] = *(const int4*)(qrow + kc * 32 + ko);
            fq[kc].q[1] = *(const int4*)(qrow + kc * 32 + 8 + ko);
        }
    }

    // ---- S = Q . K^T  (16 j-tiles x 4 K-chunks = 64 WMMAs) ----
    v8f s[16];
    const v8f vzero = {0.f, 0.f, 0.f, 0.f, 0.f, 0.f, 0.f, 0.f};
#pragma unroll
    for (int jt = 0; jt < 16; ++jt) s[jt] = vzero;

#pragma unroll
    for (int jt = 0; jt < 16; ++jt) {
        const _Float16* krow = Kb + (size_t)(jt * 16 + lm) * DK + lh * 16;
#pragma unroll
        for (int kc = 0; kc < 4; ++kc) {
            H16 fk;
            fk.q[0] = *(const int4*)(krow + kc * 32);
            fk.q[1] = *(const int4*)(krow + kc * 32 + 8);
            s[jt] = __builtin_amdgcn_wmma_f32_16x16x32_f16(
                false, fq[kc].v, false, fk.v, (short)0, s[jt], false, false);
        }
    }

    // ---- logits = S/sqrt(dk) + log-bias ----
    const float scale = 0.08838834764831845f;   // 1/sqrt(128)
#pragma unroll
    for (int jt = 0; jt < 16; ++jt) {
#pragma unroll
        for (int i = 0; i < 8; ++i) {
            const int row = qbase + lh * 8 + i;
            const int col = jt * 16 + lm;
            const float bl = (float)Bb[(size_t)row * NN + col];
            s[jt][i] = s[jt][i] * scale + bl;
        }
    }

    // ---- softmax: row max + exp + row sum (rows striped over 16 lanes) ----
    float m[8], rs[8];
#pragma unroll
    for (int i = 0; i < 8; ++i) {
        float mx = s[0][i];
#pragma unroll
        for (int jt = 1; jt < 16; ++jt) mx = fmaxf(mx, s[jt][i]);
        mx = fmaxf(mx, __shfl_xor(mx, 1, 32));
        mx = fmaxf(mx, __shfl_xor(mx, 2, 32));
        mx = fmaxf(mx, __shfl_xor(mx, 4, 32));
        mx = fmaxf(mx, __shfl_xor(mx, 8, 32));
        m[i] = mx;
    }
#pragma unroll
    for (int i = 0; i < 8; ++i) {
        float sm = 0.0f;
        const int r = lh * 8 + i;
#pragma unroll
        for (int jt = 0; jt < 16; ++jt) {
            const float p = __expf(s[jt][i] - m[i]);
            sm += p;
            pl[wave][r][jt * 16 + lm] = (_Float16)p;
        }
        sm += __shfl_xor(sm, 1, 32);
        sm += __shfl_xor(sm, 2, 32);
        sm += __shfl_xor(sm, 4, 32);
        sm += __shfl_xor(sm, 8, 32);
        rs[i] = sm;
    }
    __syncthreads();   // P visible (per-wave region; barrier is belt+braces)

    // ---- O = P . V  (8 d-tiles x 8 key-chunks = 64 WMMAs) ----
#pragma unroll
    for (int dt = 0; dt < 8; ++dt) {
        v8f o = vzero;
        const _Float16* vrow = Vb + (size_t)(dt * 16 + lm) * NN + lh * 16;
#pragma unroll
        for (int kc = 0; kc < 8; ++kc) {
            H16 fp, fv;
            const int ko = lh * 16;
            fp.q[0] = *(const int4*)&pl[wave][lm][kc * 32 + ko];
            fp.q[1] = *(const int4*)&pl[wave][lm][kc * 32 + 8 + ko];
            fv.q[0] = *(const int4*)(vrow + kc * 32);
            fv.q[1] = *(const int4*)(vrow + kc * 32 + 8);
            o = __builtin_amdgcn_wmma_f32_16x16x32_f16(
                false, fp.v, false, fv.v, (short)0, o, false, false);
        }
#pragma unroll
        for (int i = 0; i < 8; ++i) {
            const int row = qbase + lh * 8 + i;
            Oout[((size_t)b * NN + row) * DD + h * DK + dt * 16 + lm] =
                (_Float16)(o[i] / rs[i]);
        }
    }
}

// ---------------------------------------------------------------------------
// Host launcher
// ---------------------------------------------------------------------------
extern "C" void kernel_launch(void* const* d_in, const int* in_sizes, int n_in,
                              void* d_out, int out_size, void* d_ws, size_t ws_size,
                              hipStream_t stream) {
    (void)in_sizes; (void)n_in; (void)out_size; (void)ws_size;

    const float* inQ = (const float*)d_in[0];
    const float* inK = (const float*)d_in[1];
    const float* inV = (const float*)d_in[2];
    const float* box = (const float*)d_in[3];
    const float* Wq  = (const float*)d_in[4];
    const float* bq  = (const float*)d_in[5];
    const float* Wk  = (const float*)d_in[6];
    const float* bk  = (const float*)d_in[7];
    const float* Wv  = (const float*)d_in[8];
    const float* bv  = (const float*)d_in[9];
    const float* Wg  = (const float*)d_in[10];
    const float* bg  = (const float*)d_in[11];
    const float* Wo  = (const float*)d_in[12];
    const float* bo  = (const float*)d_in[13];

    char* ws = (char*)d_ws;
    _Float16* Wt    = (_Float16*)(ws);                      // 4 x 1M f16 = 8 MB
    _Float16* QhT   = (_Float16*)(ws + ((size_t)8  << 20)); // 8 MB
    _Float16* KhT   = (_Float16*)(ws + ((size_t)16 << 20)); // 8 MB
    _Float16* VtT   = (_Float16*)(ws + ((size_t)24 << 20)); // 8 MB
    _Float16* biasL = (_Float16*)(ws + ((size_t)32 << 20)); // 16 MB
    _Float16* attn  = (_Float16*)(ws + ((size_t)48 << 20)); // 8 MB

    prep_weights<<<dim3(4096, 4), 256, 0, stream>>>(Wq, Wk, Wv, Wo, Wt);

    const dim3 gGemm(32, 8);
    gemm_proj<<<gGemm, 256, 0, stream>>>(inQ, nullptr, Wt,                bq, QhT, nullptr, 0);
    gemm_proj<<<gGemm, 256, 0, stream>>>(inK, nullptr, Wt + (1u << 20),   bk, KhT, nullptr, 1);
    gemm_proj<<<gGemm, 256, 0, stream>>>(inV, nullptr, Wt + (2u << 20),   bv, VtT, nullptr, 2);

    bias_kernel<<<dim3(256, 16), 256, 0, stream>>>(box, Wg, bg, biasL);

    attn_kernel<<<512, 128, 0, stream>>>(QhT, KhT, VtT, biasL, attn);

    gemm_proj<<<gGemm, 256, 0, stream>>>(nullptr, attn, Wt + (3u << 20),  bo,
                                         nullptr, (float*)d_out, 3);
}
